// MultiHeadDistanceLayer_4801773437266
// MI455X (gfx1250) — matrix operand order repaired and checked by reading, not compiled, not run on hardware
//
#include <hip/hip_runtime.h>
#include <hip/hip_bf16.h>
#include <math.h>

// Problem constants (from reference)
#define BATCH 4
#define SEQL  2048
#define CIN   64
#define NHEAD 8
#define HDIM  32
#define MD    128
#define WFULL 257      // 2*MD+1
#define WPAD  272      // 17 * 16
#define WSM   64
#define MROWS 8192     // BATCH*SEQL

typedef __attribute__((ext_vector_type(16))) _Float16 v16h;
typedef __attribute__((ext_vector_type(8)))  _Float16 v8h;
typedef __attribute__((ext_vector_type(8)))  float    v8f;

// A-matrix (16x32 f16): lane m (both halves) holds row m.
// Lanes 0-15 hold K {0..15}; lanes 16-31 hold K {16..31}.
// Slot i: K = i + 16*group  => two contiguous 8-half runs.
__device__ __forceinline__ v16h load_a_frag(const _Float16* __restrict__ p, int group) {
    v8h lo = *(const v8h*)(p + group * 16);       // K = 16g .. 16g+7
    v8h hi = *(const v8h*)(p + 8 + group * 16);   // K = 16g+8 .. 16g+15
    v16h r;
    #pragma unroll
    for (int i = 0; i < 8; ++i) { r[i] = lo[i]; r[i + 8] = hi[i]; }
    return r;
}

// ---------------------------------------------------------------------------
// Kernel: zero output accumulator
// ---------------------------------------------------------------------------
__global__ void zero_out_kernel(float* __restrict__ out, int n) {
    int i = blockIdx.x * blockDim.x + threadIdx.x;
    if (i < n) out[i] = 0.0f;
}

// ---------------------------------------------------------------------------
// Kernel: convert Wq/Wk (64x256 f32, [k][n]) -> f16 transposed wt[z][n][k]
// so projection B fragments are contiguous per lane.
// ---------------------------------------------------------------------------
__global__ void wcvt_kernel(const float* __restrict__ Wq, const float* __restrict__ Wk,
                            _Float16* __restrict__ wt)
{
    int tid = blockIdx.x * blockDim.x + threadIdx.x;   // 2*256*64
    if (tid >= 2 * 256 * CIN) return;
    int z = tid >> 14;              // 16384 per matrix
    int rem = tid & 16383;
    int n = rem >> 6, k = rem & 63;
    const float* W = z ? Wk : Wq;
    wt[tid] = (_Float16)W[k * 256 + n];
}

// ---------------------------------------------------------------------------
// Kernel: QK projection via WMMA.  grid = (512, 16, 2), block = 32 (1 wave).
//   z==0: q path -> writes qu = q+u_pe, qv = q+v_pe (f16, layout [h][b][l][d])
//   z==1: k path -> writes kf (f16, layout [h][b][l][d])
// ---------------------------------------------------------------------------
__global__ __launch_bounds__(32)
void proj_kernel(const float* __restrict__ x,
                 const _Float16* __restrict__ wt,
                 const float* __restrict__ bq, const float* __restrict__ bk,
                 const float* __restrict__ u_pe, const float* __restrict__ v_pe,
                 _Float16* __restrict__ qu, _Float16* __restrict__ qv,
                 _Float16* __restrict__ kf)
{
    const int lane = threadIdx.x;
    const int r0 = blockIdx.x * 16;            // row tile over B*L
    const int n0 = blockIdx.y * 16;            // col tile over H*D = 256
    const bool isK = (blockIdx.z == 1);
    const float* bias = isK ? bk : bq;
    const int group = (lane >> 4) & 1;

    // A fragments from x (f32 -> f16); K runs are contiguous per lane.
    const float* xp = x + (size_t)(r0 + (lane & 15)) * CIN;
    v16h a0, a1;
    #pragma unroll
    for (int i = 0; i < 8; ++i) {
        a0[i]     = (_Float16)xp[group * 16 + i];          // K 16g..16g+7
        a0[i + 8] = (_Float16)xp[8 + group * 16 + i];      // K 16g+8..
        a1[i]     = (_Float16)xp[32 + group * 16 + i];
        a1[i + 8] = (_Float16)xp[40 + group * 16 + i];
    }
    // B fragments from transposed f16 weights: lane = column, K contiguous.
    const int bcol  = n0 + (lane & 15);
    const int kbase = (lane & 16) ? 16 : 0;
    const _Float16* wp = wt + (size_t)(isK ? 16384 : 0) + (size_t)bcol * CIN;
    v16h b0 = *(const v16h*)(wp + kbase);
    v16h b1 = *(const v16h*)(wp + 32 + kbase);

    v8f acc = {};
    acc = __builtin_amdgcn_wmma_f32_16x16x32_f16(false, a0, false, b0, (short)0, acc, false, false);
    acc = __builtin_amdgcn_wmma_f32_16x16x32_f16(false, a1, false, b1, (short)0, acc, false, false);

    const int cc = n0 + (lane & 15);
    const int h = cc >> 5, d = cc & 31;
    #pragma unroll
    for (int r = 0; r < 8; ++r) {
        int m = r + ((lane & 16) ? 8 : 0);
        int R = r0 + m;
        int b = R >> 11;                       // SEQL = 2048
        int l = R & 2047;
        float val = acc[r] + bias[cc];
        size_t idx = ((size_t)(h * BATCH + b) * SEQL + l) * HDIM + d;
        if (isK) {
            kf[idx] = (_Float16)val;
        } else {
            qu[idx] = (_Float16)(val + u_pe[h * HDIM + d]);
            qv[idx] = (_Float16)(val + v_pe[h * HDIM + d]);
        }
    }
}

// ---------------------------------------------------------------------------
// Kernel: sigmoid value gate  v[h][b][l] = sigmoid(x[b,l,:] @ Wv[:,h])
// ---------------------------------------------------------------------------
__global__ void vgate_kernel(const float* __restrict__ x,
                             const float* __restrict__ Wv,
                             float* __restrict__ vbuf)
{
    int tid = blockIdx.x * blockDim.x + threadIdx.x;   // B*L*H
    if (tid >= BATCH * SEQL * NHEAD) return;
    int h  = tid & (NHEAD - 1);
    int bl = tid >> 3;
    int b = bl >> 11, l = bl & 2047;
    float s = 0.0f;
    #pragma unroll 8
    for (int c = 0; c < CIN; ++c) s += x[bl * CIN + c] * Wv[c * NHEAD + h];
    vbuf[(size_t)(h * BATCH + b) * SEQL + l] = 1.0f / (1.0f + __expf(-s));
}

// ---------------------------------------------------------------------------
// Kernel: smooth_pe  (H,D,64) --bilinear half-pixel--> (H,257,D) f16 (note
// TRANSPOSED layout [h][w][d], stride D), padded to 272 w-columns with zeros,
// so attention-PE B fragments are one contiguous v16h per lane.
// ---------------------------------------------------------------------------
__global__ void spe_kernel(const float* __restrict__ dpe, _Float16* __restrict__ spe)
{
    int tid = blockIdx.x * blockDim.x + threadIdx.x;   // H*WPAD*D
    if (tid >= NHEAD * WPAD * HDIM) return;
    int d = tid & (HDIM - 1);
    int w = (tid >> 5) % WPAD;
    int h = tid / (WPAD * HDIM);
    float val = 0.0f;
    if (w < WFULL) {
        float xf = (w + 0.5f) * ((float)WSM / (float)WFULL) - 0.5f;
        float fl = floorf(xf);
        int   i0 = (int)fl;
        float t  = xf - fl;
        int i0c = i0 < 0 ? 0 : (i0 > WSM - 1 ? WSM - 1 : i0);
        int i1c = (i0 + 1) < 0 ? 0 : ((i0 + 1) > WSM - 1 ? WSM - 1 : i0 + 1);
        const float* src = dpe + (size_t)(h * HDIM + d) * WSM;
        val = src[i0c] * (1.0f - t) + src[i1c] * t;
    }
    spe[tid] = (_Float16)val;
}

// ---------------------------------------------------------------------------
// Kernel: fused scores (band WMMA + PE WMMA) -> softmax -> gate -> reduce.
// grid = (L/16, B, H), block = 32 (1 wave).
// ---------------------------------------------------------------------------
__global__ __launch_bounds__(32)
void att_kernel(const _Float16* __restrict__ qu, const _Float16* __restrict__ qv,
                const _Float16* __restrict__ kf, const float* __restrict__ vbuf,
                const _Float16* __restrict__ spe, float* __restrict__ out)
{
    __shared__ float scores[16][WPAD];

    const int lane = threadIdx.x;
    const int n0 = blockIdx.x * 16;
    const int b  = blockIdx.y;
    const int h  = blockIdx.z;
    const size_t base = (size_t)(h * BATCH + b) * SEQL;
    const int group = (lane >> 4) & 1;
    const int kbase = group ? 16 : 0;

    // ---------------- Phase 1: PE term  (q+v_pe) @ smooth_pe -------------
    v16h a_pe = load_a_frag(qv + (base + n0 + (lane & 15)) * HDIM, group);
    const _Float16* speh = spe + (size_t)h * WPAD * HDIM;
    for (int t = 0; t < 17; ++t) {
        int w = t * 16 + (lane & 15);
        v16h bf = *(const v16h*)(speh + (size_t)w * HDIM + kbase);
        v8f acc = {};
        acc = __builtin_amdgcn_wmma_f32_16x16x32_f16(false, a_pe, false, bf, (short)0, acc, false, false);
        #pragma unroll
        for (int r = 0; r < 8; ++r) {
            int m = r + ((lane & 16) ? 8 : 0);
            scores[m][t * 16 + (lane & 15)] = acc[r];
        }
    }
    __syncthreads();

    // ---------------- Phase 2: band term  <q_u[j+128-w], k[j]> -----------
    const int j0 = SEQL - 1 - n0;              // j for row m=0; j(m) = j0 - m
    v16h a_k = load_a_frag(kf + (base + j0 - (lane & 15)) * HDIM, group);
    const int qstart = j0 - 143;               // 272 q-columns cover the band
    for (int t = 0; t < 17; ++t) {
        int q = qstart + t * 16 + (lane & 15);
        // prefetch next tile's q-columns (global_prefetch_b8)
        if (t < 16 && (q + 16) >= 0 && (q + 16) < SEQL)
            __builtin_prefetch(qu + (base + q + 16) * HDIM + kbase, 0, 0);
        v16h bf = {};
        if (q >= 0 && q < SEQL)
            bf = *(const v16h*)(qu + (base + q) * HDIM + kbase);
        v8f acc = {};
        acc = __builtin_amdgcn_wmma_f32_16x16x32_f16(false, a_k, false, bf, (short)0, acc, false, false);
        int c = lane & 15;
        int qc = qstart + t * 16 + c;
        #pragma unroll
        for (int r = 0; r < 8; ++r) {
            int m = r + ((lane & 16) ? 8 : 0);
            int w = (j0 - m) + MD - qc;        // anti-diagonal scatter
            if (w >= 0 && w < WFULL) scores[m][w] += acc[r];
        }
    }
    __syncthreads();

    // ---------------- Phase 3: softmax, gate, column reduce --------------
    const float scale = 0.17677669529663687f;  // 1/sqrt(32)
    for (int m = 0; m < 16; ++m) {
        float mx = -3.4e38f;
        for (int w = lane; w < WFULL; w += 32) mx = fmaxf(mx, scores[m][w]);
        #pragma unroll
        for (int off = 16; off; off >>= 1) mx = fmaxf(mx, __shfl_xor(mx, off));
        mx *= scale;
        float sum = 0.0f;
        for (int w = lane; w < WFULL; w += 32) {
            float e = __expf(scores[m][w] * scale - mx);
            scores[m][w] = e;
            sum += e;
        }
        #pragma unroll
        for (int off = 16; off; off >>= 1) sum += __shfl_xor(sum, off);
        float g = vbuf[base + (n0 + m)] / sum;
        for (int w = lane; w < WFULL; w += 32) scores[m][w] *= g;
    }
    __syncthreads();
    for (int w = lane; w < WFULL; w += 32) {
        float acc = 0.0f;
        #pragma unroll
        for (int m = 0; m < 16; ++m) acc += scores[m][w];
        atomicAdd(&out[(size_t)(b * WFULL + w) * NHEAD + h], acc);
    }
}

// ---------------------------------------------------------------------------
// Launch
// ---------------------------------------------------------------------------
extern "C" void kernel_launch(void* const* d_in, const int* in_sizes, int n_in,
                              void* d_out, int out_size, void* d_ws, size_t ws_size,
                              hipStream_t stream) {
    const float* x   = (const float*)d_in[0];
    const float* Wq  = (const float*)d_in[1];
    const float* bq  = (const float*)d_in[2];
    const float* Wk  = (const float*)d_in[3];
    const float* bk  = (const float*)d_in[4];
    const float* Wv  = (const float*)d_in[5];
    const float* dpe = (const float*)d_in[6];
    const float* upe = (const float*)d_in[7];
    const float* vpe = (const float*)d_in[8];
    float* out = (float*)d_out;

    // Workspace layout (all offsets multiples of 256B)
    char* ws = (char*)d_ws;
    const size_t QKSZ  = (size_t)NHEAD * BATCH * SEQL * HDIM * sizeof(_Float16); // 4 MB
    const size_t VSZ   = (size_t)NHEAD * BATCH * SEQL * sizeof(float);           // 256 KB
    const size_t SPESZ = (size_t)NHEAD * WPAD * HDIM * sizeof(_Float16);         // 136 KB
    _Float16* qu   = (_Float16*)(ws);
    _Float16* qv   = (_Float16*)(ws + QKSZ);
    _Float16* kf   = (_Float16*)(ws + 2 * QKSZ);
    float*    vbuf = (float*)   (ws + 3 * QKSZ);
    _Float16* spe  = (_Float16*)(ws + 3 * QKSZ + VSZ);
    _Float16* wt   = (_Float16*)(ws + 3 * QKSZ + VSZ + SPESZ);

    zero_out_kernel<<<(BATCH * WFULL * NHEAD + 255) / 256, 256, 0, stream>>>(out, BATCH * WFULL * NHEAD);
    wcvt_kernel<<<(2 * 256 * CIN + 255) / 256, 256, 0, stream>>>(Wq, Wk, wt);
    spe_kernel<<<(NHEAD * WPAD * HDIM + 255) / 256, 256, 0, stream>>>(dpe, spe);
    vgate_kernel<<<(BATCH * SEQL * NHEAD + 255) / 256, 256, 0, stream>>>(x, Wv, vbuf);

    dim3 gproj(MROWS / 16, 256 / 16, 2);
    proj_kernel<<<gproj, 32, 0, stream>>>(x, wt, bq, bk, upe, vpe, qu, qv, kf);

    att_kernel<<<dim3(SEQL / 16, BATCH, NHEAD), 32, 0, stream>>>(qu, qv, kf, vbuf, spe, out);
}
